// MultiHeadAttention_8083128451184
// MI455X (gfx1250) — matrix operand order repaired and checked by reading, not compiled, not run on hardware
//
#include <hip/hip_runtime.h>
#include <hip/hip_bf16.h>

// ---------------------------------------------------------------------------
// MHA forward for MI455X (gfx1250, wave32, WMMA).
// B=4, T=2048, D=1024, H=16, HD=64. All GEMMs run on v_wmma_f32_16x16x32_bf16.
// ---------------------------------------------------------------------------

typedef __bf16 bf16;
typedef __attribute__((ext_vector_type(16))) __bf16 v16bf;
typedef __attribute__((ext_vector_type(8)))  float  v8f;

union FragBF { v16bf v; uint4 u[2]; };

constexpr int kB  = 4;
constexpr int kT  = 2048;
constexpr int kD  = 1024;
constexpr int kH  = 16;
constexpr int kHD = 64;
constexpr int kM  = kB * kT;   // 8192 tokens

__device__ __forceinline__ v8f zero8() {
    v8f z = {0.f, 0.f, 0.f, 0.f, 0.f, 0.f, 0.f, 0.f};
    return z;
}

// A-fragment, 16x32 bf16, row-major source with row stride lda (elements).
// ISA layout: lanes 0-15 row=l, K={0..7,16..23}; lanes 16-31 row=l-16, K={8..15,24..31}.
__device__ __forceinline__ v16bf load_fragA(const bf16* base, int lda) {
    const int lane = threadIdx.x & 31;
    const bf16* p = base + (size_t)(lane & 15) * lda + ((lane < 16) ? 0 : 8);
    FragBF f;
    f.u[0] = *reinterpret_cast<const uint4*>(p);
    f.u[1] = *reinterpret_cast<const uint4*>(p + 16);
    return f.v;
}

// B-fragment, 32x16 bf16. Source is "Bt": row n holds the 32 K-contiguous values.
// ISA layout: lanes 0-15 col=l, K={0..7,16..23}; lanes 16-31 col=l-16, K={8..15,24..31}.
__device__ __forceinline__ v16bf load_fragB(const bf16* base, int ldb) {
    const int lane = threadIdx.x & 31;
    const bf16* p = base + (size_t)(lane & 15) * ldb + ((lane < 16) ? 0 : 8);
    FragBF f;
    f.u[0] = *reinterpret_cast<const uint4*>(p);
    f.u[1] = *reinterpret_cast<const uint4*>(p + 16);
    return f.v;
}

__device__ __forceinline__ v8f wmma_bf16(v16bf a, v16bf b, v8f c) {
    return __builtin_amdgcn_wmma_f32_16x16x32_bf16(
        /*neg_a=*/false, a, /*neg_b=*/false, b,
        /*c_mod=*/(short)0, c, /*reuse_a=*/false, /*reuse_b=*/false);
}

// 16-lane-group reductions (xor masks < 16 stay within each half-wave group,
// matching the C-matrix layout where a row lives on lanes 0-15 or 16-31).
__device__ __forceinline__ float half16_max(float x) {
    x = fmaxf(x, __shfl_xor(x, 1, 32));
    x = fmaxf(x, __shfl_xor(x, 2, 32));
    x = fmaxf(x, __shfl_xor(x, 4, 32));
    x = fmaxf(x, __shfl_xor(x, 8, 32));
    return x;
}
__device__ __forceinline__ float half16_sum(float x) {
    x += __shfl_xor(x, 1, 32);
    x += __shfl_xor(x, 2, 32);
    x += __shfl_xor(x, 4, 32);
    x += __shfl_xor(x, 8, 32);
    return x;
}

// ------------------------------- converters --------------------------------

__global__ void k_f32_to_bf16(const float* __restrict__ src,
                              bf16* __restrict__ dst, int n) {
    int i = blockIdx.x * blockDim.x + threadIdx.x;
    if (i < n) dst[i] = (bf16)src[i];
}

// W[k][n] (DxD f32) -> Wt[n][k] (bf16): B-operand wants K contiguous per row n.
__global__ void k_transpose_to_bf16(const float* __restrict__ src,
                                    bf16* __restrict__ dst) {
    int i = blockIdx.x * blockDim.x + threadIdx.x;  // i = k*D + n
    int k = i >> 10;
    int n = i & (kD - 1);
    dst[(size_t)n * kD + k] = (bf16)src[i];
}

// --------------------------------- GEMM ------------------------------------
// C[M,N] = A[M,K] @ B, with B given as Bt[N,K] (both bf16), f32 accumulate.
// OUT_MODE 0: f32 row-major; 1: bf16 row-major; 2: bf16 stored as V^T [B,H,HD,T].
// Block: 256 threads = 8 waves as 4(m) x 2(n); block tile 128x64; wave tile 32x32.

template <int OUT_MODE>
__global__ __launch_bounds__(256) void k_gemm_bf16(const bf16* __restrict__ A,
                                                   const bf16* __restrict__ Bt,
                                                   void* __restrict__ Cout) {
    const int lane = threadIdx.x & 31;
    const int wave = threadIdx.x >> 5;
    const int wm = wave & 3;
    const int wn = wave >> 2;
    const int m0 = blockIdx.y * 128 + wm * 32;
    const int n0 = blockIdx.x * 64 + wn * 32;
    constexpr int K = kD;

    v8f acc[2][2];
    for (int i = 0; i < 2; ++i)
        for (int j = 0; j < 2; ++j) acc[i][j] = zero8();

    const bf16* pa0 = A + (size_t)m0 * K;
    const bf16* pb0 = Bt + (size_t)n0 * K;

    for (int k0 = 0; k0 < K; k0 += 32) {
        if (k0 + 64 < K) {  // global_prefetch_b8 of the next K panel
            __builtin_prefetch(pa0 + (size_t)(lane & 15) * K + k0 + 64, 0, 1);
            __builtin_prefetch(pb0 + (size_t)(lane & 15) * K + k0 + 64, 0, 1);
        }
        v16bf a0 = load_fragA(pa0 + k0, K);
        v16bf a1 = load_fragA(pa0 + (size_t)16 * K + k0, K);
        v16bf b0 = load_fragB(pb0 + k0, K);
        v16bf b1 = load_fragB(pb0 + (size_t)16 * K + k0, K);
        acc[0][0] = wmma_bf16(a0, b0, acc[0][0]);
        acc[0][1] = wmma_bf16(a0, b1, acc[0][1]);
        acc[1][0] = wmma_bf16(a1, b0, acc[1][0]);
        acc[1][1] = wmma_bf16(a1, b1, acc[1][1]);
    }

    const int mo = (lane < 16) ? 0 : 8;
    const int nn = lane & 15;
    for (int i = 0; i < 2; ++i)
        for (int j = 0; j < 2; ++j)
            for (int r = 0; r < 8; ++r) {
                const int gm = m0 + i * 16 + mo + r;
                const int gn = n0 + j * 16 + nn;
                const float v = acc[i][j][r];
                if (OUT_MODE == 0) {
                    ((float*)Cout)[(size_t)gm * kD + gn] = v;
                } else if (OUT_MODE == 1) {
                    ((bf16*)Cout)[(size_t)gm * kD + gn] = (bf16)v;
                } else {
                    const int bb = gm >> 11;           // gm / T
                    const int t  = gm & (kT - 1);
                    const int hh = gn >> 6;            // gn / HD
                    const int d  = gn & (kHD - 1);
                    ((bf16*)Cout)[(((size_t)bb * kH + hh) * kHD + d) * kT + t] =
                        (bf16)v;
                }
            }
}

// ----------------------------- flash attention -----------------------------
// One wave owns 16 query rows of one (b,h). Streams causal KV in 32-wide tiles.
// Q,K are bf16 [B*T, D]; V is bf16 transposed [B,H,HD,T]; Ctx out bf16 [B*T,D].

__global__ __launch_bounds__(256) void k_flash(const bf16* __restrict__ Q,
                                               const bf16* __restrict__ Kb,
                                               const bf16* __restrict__ Vt,
                                               bf16* __restrict__ Ctx) {
    __shared__ bf16 lds[8][16 * 32];  // per-wave P staging (C-layout -> A-layout)

    const int lane = threadIdx.x & 31;
    const int wave = threadIdx.x >> 5;
    const int task = blockIdx.x * 8 + wave;        // B*H*(T/16) = 8192 tasks
    const int QT = kT / 16;
    const int qt = task % QT;
    const int hb = task / QT;
    const int h = hb % kH;
    const int b = hb / kH;
    const int qbase = qt * 16;

    const size_t rowQ = ((size_t)b * kT + qbase) * kD + h * kHD;
    const v16bf aq0 = load_fragA(Q + rowQ, kD);        // d = 0..31
    const v16bf aq1 = load_fragA(Q + rowQ + 32, kD);   // d = 32..63

    v8f ctx[4];
    float m_run[8], l_run[8];
    for (int nt = 0; nt < 4; ++nt) ctx[nt] = zero8();
    for (int r = 0; r < 8; ++r) { m_run[r] = -3.0e38f; l_run[r] = 0.f; }

    const int mo = (lane < 16) ? 0 : 8;
    const int nn = lane & 15;
    bf16* ldsw = lds[wave];

    for (int t0 = 0; t0 < qbase + 16; t0 += 32) {
        // ---- S = Q * K^T (16 q-rows x 32 kv-cols), scale 1/sqrt(HD) ----
        v8f s[2];
        for (int nh = 0; nh < 2; ++nh) {
            const size_t rowK = ((size_t)b * kT + t0 + nh * 16) * kD + h * kHD;
            v16bf bk0 = load_fragB(Kb + rowK, kD);       // d = 0..31
            v16bf bk1 = load_fragB(Kb + rowK + 32, kD);  // d = 32..63
            v8f c = zero8();
            c = wmma_bf16(aq0, bk0, c);
            c = wmma_bf16(aq1, bk1, c);
            s[nh] = c;
        }
        // ---- causal mask + scale ----
        for (int nh = 0; nh < 2; ++nh) {
            const int kg = t0 + nh * 16 + nn;
            for (int r = 0; r < 8; ++r) {
                const float v = s[nh][r] * 0.125f;  // 1/sqrt(64)
                const int qg = qbase + mo + r;
                s[nh][r] = (kg <= qg) ? v : -1.0e30f;
            }
        }
        // ---- online softmax (per query row r within this lane group) ----
        for (int r = 0; r < 8; ++r) {
            float rm = half16_max(fmaxf(s[0][r], s[1][r]));
            const float mnew = fmaxf(m_run[r], rm);
            const float alpha = __expf(m_run[r] - mnew);
            const float p0 = __expf(s[0][r] - mnew);
            const float p1 = __expf(s[1][r] - mnew);
            s[0][r] = p0;
            s[1][r] = p1;
            const float ls = half16_sum(p0 + p1);
            l_run[r] = l_run[r] * alpha + ls;
            m_run[r] = mnew;
            for (int nt = 0; nt < 4; ++nt) ctx[nt][r] *= alpha;
        }
        // ---- P (C layout) -> LDS -> A-fragment layout ----
        for (int nh = 0; nh < 2; ++nh)
            for (int r = 0; r < 8; ++r)
                ldsw[(mo + r) * 32 + nh * 16 + nn] = (bf16)s[nh][r];
        asm volatile("s_wait_dscnt 0" ::: "memory");  // intra-wave LDS RAW
        const bf16* lp = ldsw + (lane & 15) * 32 + ((lane < 16) ? 0 : 8);
        FragBF f;
        f.u[0] = *reinterpret_cast<const uint4*>(lp);
        f.u[1] = *reinterpret_cast<const uint4*>(lp + 16);
        const v16bf pa = f.v;
        // ---- ctx += P * V  (V^T rows are d, contiguous in t) ----
        for (int nt = 0; nt < 4; ++nt) {
            const bf16* vb =
                Vt + (((size_t)b * kH + h) * kHD + nt * 16) * kT + t0;
            v16bf bv = load_fragB(vb, kT);
            ctx[nt] = wmma_bf16(pa, bv, ctx[nt]);
        }
    }

    // ---- finalize: divide by row sums, write merged-head context ----
    for (int r = 0; r < 8; ++r) {
        const float inv = 1.0f / l_run[r];
        for (int nt = 0; nt < 4; ++nt) ctx[nt][r] *= inv;
    }
    for (int nt = 0; nt < 4; ++nt)
        for (int r = 0; r < 8; ++r) {
            const int qg = qbase + mo + r;
            Ctx[((size_t)b * kT + qg) * kD + h * kHD + nt * 16 + nn] =
                (bf16)ctx[nt][r];
        }
}

// ------------------------------ launcher -----------------------------------

extern "C" void kernel_launch(void* const* d_in, const int* in_sizes, int n_in,
                              void* d_out, int out_size, void* d_ws,
                              size_t ws_size, hipStream_t stream) {
    const float* x  = (const float*)d_in[0];
    const float* Wq = (const float*)d_in[1];
    const float* Wk = (const float*)d_in[2];
    const float* Wv = (const float*)d_in[3];
    const float* Wo = (const float*)d_in[4];
    float* out = (float*)d_out;

    // Workspace layout (bytes): needs 88 MB.
    char* ws = (char*)d_ws;
    bf16* xb  = (bf16*)(ws);                       // 16 MB  x in bf16
    bf16* WqT = (bf16*)(ws + (16u << 20));         //  2 MB  Wq^T bf16
    bf16* WkT = (bf16*)(ws + (18u << 20));
    bf16* WvT = (bf16*)(ws + (20u << 20));
    bf16* WoT = (bf16*)(ws + (22u << 20));
    bf16* Qb  = (bf16*)(ws + (24u << 20));         // 16 MB  Q bf16 [B*T,D]
    bf16* Kb  = (bf16*)(ws + (40u << 20));         // 16 MB  K bf16 [B*T,D]
    bf16* Vt  = (bf16*)(ws + (56u << 20));         // 16 MB  V^T bf16 [B,H,HD,T]
    bf16* Cx  = (bf16*)(ws + (72u << 20));         // 16 MB  ctx bf16 [B*T,D]

    const int nx = kM * kD;
    k_f32_to_bf16<<<nx / 256, 256, 0, stream>>>(x, xb, nx);
    const int nw = kD * kD;
    k_transpose_to_bf16<<<nw / 256, 256, 0, stream>>>(Wq, WqT);
    k_transpose_to_bf16<<<nw / 256, 256, 0, stream>>>(Wk, WkT);
    k_transpose_to_bf16<<<nw / 256, 256, 0, stream>>>(Wv, WvT);
    k_transpose_to_bf16<<<nw / 256, 256, 0, stream>>>(Wo, WoT);

    dim3 g(kD / 64, kM / 128);  // (16, 64)
    k_gemm_bf16<1><<<g, 256, 0, stream>>>(xb, WqT, Qb);
    k_gemm_bf16<1><<<g, 256, 0, stream>>>(xb, WkT, Kb);
    k_gemm_bf16<2><<<g, 256, 0, stream>>>(xb, WvT, Vt);

    k_flash<<<(kB * kH * (kT / 16)) / 8, 256, 0, stream>>>(Qb, Kb, Vt, Cx);

    k_gemm_bf16<0><<<g, 256, 0, stream>>>(Cx, WoT, out);
}
